// AttentionLayer_57088705298643
// MI455X (gfx1250) — matrix operand order repaired and checked by reading, not compiled, not run on hardware
//
#include <hip/hip_runtime.h>

#define NROWS 10000
#define DFEAT 128
#define NPAD  10016            // NROWS rounded up to a multiple of 32
#define KTILES (NPAD / 32)     // 313

typedef __attribute__((ext_vector_type(16))) __bf16 v16bf;
typedef __attribute__((ext_vector_type(8)))  float  v8f;
typedef float f4v __attribute__((ext_vector_type(4)));

union ABfrag { v16bf v; uint4 u[2]; };

__device__ __forceinline__ unsigned short f2bf(float f) {
    unsigned int x = __float_as_uint(f);
    x += 0x7fffu + ((x >> 16) & 1u);   // round-to-nearest-even
    return (unsigned short)(x >> 16);
}

// Kernel 1: score[j] = inputs[j,:] . Hv ; inputsT[d][j] = bf16(inputs[j][d])
// (zero-filled for j in [NROWS, NPAD))
__global__ void __launch_bounds__(256)
prep_kernel(const float* __restrict__ inputs,
            const float* __restrict__ Hv,
            float* __restrict__ score,
            unsigned short* __restrict__ inputsT)
{
    __shared__ __align__(16) unsigned short T[DFEAT][72]; // stride 144B (16B mult)
    const int tid = threadIdx.x;           // 256 threads
    const int j0  = blockIdx.x * 64;

    // phase 1: load [64 x 128] fp32 tile, convert to bf16, transpose into LDS
    #pragma unroll
    for (int it = 0; it < 8; ++it) {
        int idx = tid + it * 256;          // 0..2047
        int j   = idx >> 5;                // 0..63
        int d   = (idx & 31) * 4;
        f4v val = {0.f, 0.f, 0.f, 0.f};
        if (j0 + j < NROWS)
            val = *(const f4v*)(inputs + (size_t)(j0 + j) * DFEAT + d);
        T[d + 0][j] = f2bf(val[0]);
        T[d + 1][j] = f2bf(val[1]);
        T[d + 2][j] = f2bf(val[2]);
        T[d + 3][j] = f2bf(val[3]);
    }
    __syncthreads();
    // phase 2: coalesced store of transposed tile
    #pragma unroll
    for (int it = 0; it < 4; ++it) {
        int idx = tid + it * 256;          // 0..1023
        int row = idx >> 3;                // 0..127
        int c   = idx & 7;                 // 16B chunk (8 bf16)
        int j   = j0 + c * 8;
        if (j < NPAD)
            *(uint4*)(inputsT + (size_t)row * NPAD + j) = *(const uint4*)&T[row][c * 8];
    }
    // phase 3: scores in fp32 (zero in the padded tail)
    if (tid < 64) {
        int j = j0 + tid;
        if (j < NPAD) {
            float s = 0.f;
            if (j < NROWS) {
                #pragma unroll
                for (int c = 0; c < 32; ++c) {
                    f4v a = *(const f4v*)(inputs + (size_t)j * DFEAT + c * 4);
                    f4v h = *(const f4v*)(Hv + c * 4);
                    s += a[0] * h[0] + a[1] * h[1] + a[2] * h[2] + a[3] * h[3];
                }
            }
            score[j] = s;
        }
    }
}

// Kernel 2: fused masked-softmax + aggregation (flash-style single pass over adj).
// Workgroup = 128 threads = 4 waves; wave w owns rows [r0+16w, r0+16w+16), all 128 feats.
__global__ void __launch_bounds__(128)
attn_kernel(const float* __restrict__ adj,
            const float* __restrict__ score,
            const unsigned short* __restrict__ inputsT,
            float* __restrict__ out)
{
    __shared__ __align__(16) unsigned short ldsB[DFEAT][40]; // X tile [d][k], 80B stride
    __shared__ __align__(16) unsigned short ldsP[4][16][40]; // per-wave P tile [m][k]
    __shared__ float ldsF[4][16];                            // per-wave row factors/sums

    const int tid   = threadIdx.x;
    const int w     = tid >> 5;
    const int lane  = tid & 31;
    const int r     = lane >> 1;       // softmax row within wave tile (0..15)
    const int kh    = lane & 1;        // which 16-wide half of the 32-wide K tile
    const int r0    = blockIdx.x * 64 + w * 16;
    const int myrow = r0 + r;
    const bool rowok = (myrow < NROWS);

    const int an  = lane & 15;                 // fragment N index
    const int klo = (lane < 16) ? 0 : 16;      // unused; B frag K base is akb
    const int akb = (lane < 16) ? 0 : 8;       // A-frag K base
    const int fb  = (lane < 16) ? 0 : 8;       // C/D row base (M = fb + vgpr)

    float m_run = -1e30f;
    float l_run = 0.f;
    v8f acc[8];
    #pragma unroll
    for (int i = 0; i < 8; ++i) acc[i] = (v8f){0.f,0.f,0.f,0.f,0.f,0.f,0.f,0.f};

    for (int kt = 0; kt < KTILES; ++kt) {
        const int k0 = kt * 32;
        __syncthreads();   // previous tile's fragment reads done

        // stage X tile: thread t copies 32 bf16 of feature-row t (L2-resident)
        {
            const uint4* src = (const uint4*)(inputsT + (size_t)tid * NPAD + k0);
            uint4* dst = (uint4*)&ldsB[tid][0];
            dst[0] = src[0]; dst[1] = src[1]; dst[2] = src[2]; dst[3] = src[3];
        }

        // ---- online softmax over this 16x32 adj tile ----
        const int kbase = k0 + kh * 16;
        const bool kok = (kbase + 16) <= NROWS;   // NROWS % 32 == 16 -> half-granular tail
        f4v a0 = {0.f,0.f,0.f,0.f}, a1 = a0, a2 = a0, a3 = a0;
        if (rowok && kok) {
            const f4v* ap = (const f4v*)(adj + (size_t)myrow * NROWS + kbase);
            a0 = __builtin_nontemporal_load(ap + 0);   // stream adj, keep L2 for X
            a1 = __builtin_nontemporal_load(ap + 1);
            a2 = __builtin_nontemporal_load(ap + 2);
            a3 = __builtin_nontemporal_load(ap + 3);
            __builtin_prefetch((const char*)ap + 2048, 0, 1);
        }
        float av[16];
        av[0]=a0[0]; av[1]=a0[1]; av[2]=a0[2]; av[3]=a0[3];
        av[4]=a1[0]; av[5]=a1[1]; av[6]=a1[2]; av[7]=a1[3];
        av[8]=a2[0]; av[9]=a2[1]; av[10]=a2[2]; av[11]=a2[3];
        av[12]=a3[0]; av[13]=a3[1]; av[14]=a3[2]; av[15]=a3[3];

        const f4v* sp = (const f4v*)(score + kbase);   // padded + zero-filled
        f4v s0 = sp[0], s1 = sp[1], s2 = sp[2], s3 = sp[3];
        float sc[16];
        sc[0]=s0[0]; sc[1]=s0[1]; sc[2]=s0[2]; sc[3]=s0[3];
        sc[4]=s1[0]; sc[5]=s1[1]; sc[6]=s1[2]; sc[7]=s1[3];
        sc[8]=s2[0]; sc[9]=s2[1]; sc[10]=s2[2]; sc[11]=s2[3];
        sc[12]=s3[0]; sc[13]=s3[1]; sc[14]=s3[2]; sc[15]=s3[3];

        float lg[16];
        float tmax = -1e30f;
        #pragma unroll
        for (int i = 0; i < 16; ++i) {
            lg[i] = (av[i] != 0.f) ? av[i] * sc[i] : -1e30f;
            tmax  = fmaxf(tmax, lg[i]);
        }
        tmax = fmaxf(tmax, __shfl_xor(tmax, 1, 32));   // pair-combine the two k-halves
        const float m_new = fmaxf(m_run, tmax);
        const float fac   = __expf(m_run - m_new);
        float psum = 0.f;
        unsigned int pk[8];
        #pragma unroll
        for (int i = 0; i < 8; ++i) {
            float p0 = (av[2*i]   != 0.f) ? __expf(lg[2*i]   - m_new) : 0.f;
            float p1 = (av[2*i+1] != 0.f) ? __expf(lg[2*i+1] - m_new) : 0.f;
            psum += p0 + p1;
            pk[i] = (unsigned int)f2bf(p0) | ((unsigned int)f2bf(p1) << 16);
        }
        psum += __shfl_xor(psum, 1, 32);
        l_run = l_run * fac + psum;
        m_run = m_new;

        {   // store bf16 P chunk: row r, k in [kh*16, kh*16+16)
            uint4* pd = (uint4*)&ldsP[w][r][kh * 16];
            pd[0] = make_uint4(pk[0], pk[1], pk[2], pk[3]);
            pd[1] = make_uint4(pk[4], pk[5], pk[6], pk[7]);
        }
        if (kh == 0) ldsF[w][r] = fac;

        __syncthreads();   // P, factors, X tile visible

        // rescale accumulators by per-row exp(m_old - m_new)
        {
            v8f fv;
            #pragma unroll
            for (int i = 0; i < 8; ++i) fv[i] = ldsF[w][fb + i];
            #pragma unroll
            for (int t = 0; t < 8; ++t) acc[t] *= fv;
        }

        // A fragment (16x32 bf16 P tile) — shared across all 8 feature tiles
        ABfrag A;
        A.u[0] = *(const uint4*)&ldsP[w][an][akb];
        A.u[1] = *(const uint4*)&ldsP[w][an][akb + 16];

        #pragma unroll
        for (int dt = 0; dt < 8; ++dt) {
            ABfrag B;   // 32x16 bf16: column d = dt*16+an, K in A-frag order (akb..+7, akb+16..+23)
            B.u[0] = *(const uint4*)&ldsB[dt * 16 + an][akb];
            B.u[1] = *(const uint4*)&ldsB[dt * 16 + an][akb + 16];
            acc[dt] = __builtin_amdgcn_wmma_f32_16x16x32_bf16(
                false, A.v, false, B.v, (short)0, acc[dt], false, false);
        }
    }

    // epilogue: publish row sums, normalize, store fp32 output
    __syncthreads();
    if (kh == 0) ldsF[w][r] = l_run;
    __syncthreads();
    #pragma unroll
    for (int v = 0; v < 8; ++v) {
        const int orow = r0 + fb + v;
        if (orow < NROWS) {
            const float inv = 1.f / ldsF[w][fb + v];
            #pragma unroll
            for (int dt = 0; dt < 8; ++dt)
                out[(size_t)orow * DFEAT + dt * 16 + an] = acc[dt][v] * inv;
        }
    }
}

extern "C" void kernel_launch(void* const* d_in, const int* in_sizes, int n_in,
                              void* d_out, int out_size, void* d_ws, size_t ws_size,
                              hipStream_t stream) {
    (void)in_sizes; (void)n_in; (void)out_size; (void)ws_size;
    const float* inputs = (const float*)d_in[0];   // [N, D] fp32
    const float* adj    = (const float*)d_in[1];   // [N, N] fp32
    const float* Hv     = (const float*)d_in[2];   // [D, 1] fp32
    float* out          = (float*)d_out;           // [N, D] fp32

    float* score = (float*)d_ws;                                        // NPAD fp32
    unsigned short* inputsT = (unsigned short*)((char*)d_ws + 40960);   // [D][NPAD] bf16

    const int nblk = (NPAD + 63) / 64;   // 157
    prep_kernel<<<nblk, 256, 0, stream>>>(inputs, Hv, score, inputsT);
    attn_kernel<<<nblk, 128, 0, stream>>>(adj, score, inputsT, out);
}
